// rbfCKA_81612968558679
// MI455X (gfx1250) — matrix operand order repaired and checked by reading, not compiled, not run on hardware
//
#include <hip/hip_runtime.h>
#include <hip/hip_bf16.h>

typedef __attribute__((ext_vector_type(16))) __bf16 v16bf;
typedef __attribute__((ext_vector_type(8)))  __bf16 v8bf;
typedef __attribute__((ext_vector_type(8)))  float  v8f;

#define NPTS 8192
#define DIM  512
#define TILE 128
#define NB   4096
#define BIN_W 1.0f               // histogram over d2 (squared distance), range [0, 4096)
#define MEDIAN_MULT 1.0f

struct Scalars { double sumKL, sumKK, sumLL; float invA, invB; };

// ---------------- helpers ----------------

static __device__ __forceinline__ __bf16 f2bf(float f) {
    unsigned u = __float_as_uint(f);
    unsigned r = (u + 0x7FFFu + ((u >> 16) & 1u)) >> 16;
    unsigned short h = (unsigned short)r;
    return *reinterpret_cast<__bf16*>(&h);
}

// A-fragment (16x32 bf16): lane l holds M=l&15; element e -> K = 16*(e>>3) + 8*(l>>4) + (e&7)
static __device__ __forceinline__ v16bf load_a_frag(const __bf16* __restrict__ X, int row, int kb, int lh) {
    const __bf16* p = X + (size_t)row * DIM + kb + 8 * lh;
    v8bf lo = *(const v8bf*)p;          // K = kb + 8*lh + {0..7}
    v8bf hi = *(const v8bf*)(p + 16);   // K = kb + 16 + 8*lh + {0..7}
    return __builtin_shufflevector(lo, hi, 0,1,2,3,4,5,6,7,8,9,10,11,12,13,14,15);
}

// B-fragment (32x16 bf16, B[k][n] = Xj[n][k]): lane l holds N=l&15; element e -> K = 16*(e>>3) + 8*(l>>4) + (e&7)
static __device__ __forceinline__ v16bf load_b_frag(const __bf16* __restrict__ X, int row, int kb, int lh) {
    const __bf16* p = X + (size_t)row * DIM + kb + 8 * lh;
    v8bf lo = *(const v8bf*)p;          // K = kb + 8*lh + {0..7}
    v8bf hi = *(const v8bf*)(p + 16);   // K = kb + 16 + 8*lh + {0..7}
    return __builtin_shufflevector(lo, hi, 0,1,2,3,4,5,6,7,8,9,10,11,12,13,14,15);
}

// Compute 32x64 block of X * X^T for this wave: acc[mt][nt][r] = G[R0+mt*16+8*lh+r][C0+nt*16+lr]
static __device__ __forceinline__ void gram_block(const __bf16* __restrict__ X,
                                                  int R0, int C0, int lr, int lh,
                                                  v8f acc[2][4]) {
    const v8f z = {0.f,0.f,0.f,0.f,0.f,0.f,0.f,0.f};
    for (int mt = 0; mt < 2; ++mt)
        for (int nt = 0; nt < 4; ++nt)
            acc[mt][nt] = z;
    for (int kb = 0; kb < DIM; kb += 32) {
        v16bf a[2], b[4];
        for (int mt = 0; mt < 2; ++mt) a[mt] = load_a_frag(X, R0 + mt * 16 + lr, kb, lh);
        for (int nt = 0; nt < 4; ++nt) b[nt] = load_b_frag(X, C0 + nt * 16 + lr, kb, lh);
        for (int mt = 0; mt < 2; ++mt)
            for (int nt = 0; nt < 4; ++nt)
                acc[mt][nt] = __builtin_amdgcn_wmma_f32_16x16x32_bf16(
                    false, a[mt], false, b[nt], (short)0, acc[mt][nt], false, false);
    }
}

static __device__ __forceinline__ double block_reduce256(double v, double* red) {
    int tid = threadIdx.x;
    red[tid] = v;
    __syncthreads();
    for (int s = 128; s > 0; s >>= 1) {
        if (tid < s) red[tid] += red[tid + s];
        __syncthreads();
    }
    double r = red[0];
    __syncthreads();
    return r;
}

// ---------------- kernels ----------------

__global__ __launch_bounds__(256) void init_ws_kernel(unsigned* histA, unsigned* histB,
                                                      double* rowK, double* rowL, Scalars* sc) {
    int i = blockIdx.x * blockDim.x + threadIdx.x;
    if (i < NB) { histA[i] = 0u; histB[i] = 0u; }
    if (i < NPTS) { rowK[i] = 0.0; rowL[i] = 0.0; }
    if (i == 0) { sc->sumKL = 0.0; sc->sumKK = 0.0; sc->sumLL = 0.0; sc->invA = 0.f; sc->invB = 0.f; }
}

__global__ __launch_bounds__(256) void convert_kernel(const float* __restrict__ A,
                                                      const float* __restrict__ B,
                                                      __bf16* __restrict__ Abf,
                                                      __bf16* __restrict__ Bbf) {
    const int total = NPTS * DIM;
    for (int i = blockIdx.x * blockDim.x + threadIdx.x; i < total; i += gridDim.x * blockDim.x) {
        Abf[i] = f2bf(A[i]);
        Bbf[i] = f2bf(B[i]);
    }
}

__global__ __launch_bounds__(128) void rownorm_kernel(const float* __restrict__ A,
                                                      const float* __restrict__ B,
                                                      float* __restrict__ sqA,
                                                      float* __restrict__ sqB) {
    __shared__ float sm[128];
    const int row = blockIdx.x;
    const float* X = (blockIdx.y == 0) ? A : B;
    float* sq = (blockIdx.y == 0) ? sqA : sqB;
    int t = threadIdx.x;
    float s = 0.f;
    for (int j = 0; j < 4; ++j) {
        float v = X[(size_t)row * DIM + t + 128 * j];
        s += v * v;
    }
    sm[t] = s;
    __syncthreads();
    for (int st = 64; st > 0; st >>= 1) {
        if (t < st) sm[t] += sm[t + st];
        __syncthreads();
    }
    if (t == 0) sq[row] = sm[0];
}

// Histogram of SQUARED distances (median commutes with the monotone sqrt).
__global__ __launch_bounds__(256) void hist_kernel(const __bf16* __restrict__ X,
                                                   const float* __restrict__ sq,
                                                   unsigned* __restrict__ gh) {
    __shared__ unsigned lhist[NB];
    const int tid = threadIdx.x;
    for (int i = tid; i < NB; i += 256) lhist[i] = 0u;
    __syncthreads();

    const int wave = tid >> 5, lane = tid & 31;
    const int lr = lane & 15, lh = lane >> 4;
    const int R0 = blockIdx.x * TILE + (wave >> 1) * 32;
    const int C0 = blockIdx.y * TILE + (wave & 1) * 64;

    v8f acc[2][4];
    gram_block(X, R0, C0, lr, lh, acc);

    for (int mt = 0; mt < 2; ++mt) {
        for (int r = 0; r < 8; ++r) {
            const int row = R0 + mt * 16 + 8 * lh + r;
            const float sr = sq[row];
            for (int nt = 0; nt < 4; ++nt) {
                const int col = C0 + nt * 16 + lr;
                float d2 = sr + sq[col] - 2.0f * acc[mt][nt][r];
                d2 = fmaxf(d2, 0.0f);
                int bin = (int)(d2 * (1.0f / BIN_W));
                bin = min(bin, NB - 1);
                atomicAdd(&lhist[bin], 1u);
            }
        }
    }
    __syncthreads();
    for (int i = tid; i < NB; i += 256)
        if (lhist[i]) atomicAdd(&gh[i], lhist[i]);
}

__global__ void median_kernel(const unsigned* __restrict__ hA,
                              const unsigned* __restrict__ hB, Scalars* sc) {
    if (threadIdx.x != 0 || blockIdx.x != 0) return;
    const unsigned long long half = (unsigned long long)NPTS * NPTS / 2ull;
    for (int m = 0; m < 2; ++m) {
        const unsigned* h = (m == 0) ? hA : hB;
        unsigned long long cum = 0;
        int b = NB - 1;
        for (int i = 0; i < NB; ++i) {
            cum += h[i];
            if (cum >= half) { b = i; break; }
        }
        float med_d2 = ((float)b + 0.5f) * BIN_W;     // median of squared distances
        float med = sqrtf(med_d2);                    // = median distance (monotone map)
        float g = med * MEDIAN_MULT;
        float inv = 1.0f / (2.0f * g * g);
        if (m == 0) sc->invA = inv; else sc->invB = inv;
    }
}

__global__ __launch_bounds__(256) void hsic_kernel(const __bf16* __restrict__ Abf,
                                                   const __bf16* __restrict__ Bbf,
                                                   const float* __restrict__ sqA,
                                                   const float* __restrict__ sqB,
                                                   Scalars* sc,
                                                   double* __restrict__ rowK,
                                                   double* __restrict__ rowL) {
    __shared__ float rAccK[TILE];
    __shared__ float rAccL[TILE];
    __shared__ double red[256];
    const int tid = threadIdx.x;
    if (tid < TILE) { rAccK[tid] = 0.f; rAccL[tid] = 0.f; }
    __syncthreads();

    const int wave = tid >> 5, lane = tid & 31;
    const int lr = lane & 15, lh = lane >> 4;
    const int wr = wave >> 1, wc = wave & 1;
    const int R0 = blockIdx.x * TILE + wr * 32;
    const int C0 = blockIdx.y * TILE + wc * 64;

    v8f accA[2][4], accB[2][4];
    gram_block(Abf, R0, C0, lr, lh, accA);
    gram_block(Bbf, R0, C0, lr, lh, accB);

    const float invA = sc->invA, invB = sc->invB;
    float sKL = 0.f, sKK = 0.f, sLL = 0.f;

    for (int mt = 0; mt < 2; ++mt) {
        for (int r = 0; r < 8; ++r) {
            const int row = R0 + mt * 16 + 8 * lh + r;
            const int rloc = wr * 32 + mt * 16 + 8 * lh + r;
            const float srA = sqA[row], srB = sqB[row];
            float rk = 0.f, rl = 0.f;
            for (int nt = 0; nt < 4; ++nt) {
                const int col = C0 + nt * 16 + lr;
                float d2a = fmaxf(srA + sqA[col] - 2.0f * accA[mt][nt][r], 0.0f);
                float d2b = fmaxf(srB + sqB[col] - 2.0f * accB[mt][nt][r], 0.0f);
                float kv = __expf(-d2a * invA);
                float lv = __expf(-d2b * invB);
                sKL += kv * lv;
                sKK += kv * kv;
                sLL += lv * lv;
                rk += kv;
                rl += lv;
            }
            atomicAdd(&rAccK[rloc], rk);
            atomicAdd(&rAccL[rloc], rl);
        }
    }
    __syncthreads();

    double t;
    t = block_reduce256((double)sKL, red);
    if (tid == 0) atomicAdd(&sc->sumKL, t);
    t = block_reduce256((double)sKK, red);
    if (tid == 0) atomicAdd(&sc->sumKK, t);
    t = block_reduce256((double)sLL, red);
    if (tid == 0) atomicAdd(&sc->sumLL, t);

    if (tid < TILE) {
        atomicAdd(&rowK[blockIdx.x * TILE + tid], (double)rAccK[tid]);
        atomicAdd(&rowL[blockIdx.x * TILE + tid], (double)rAccL[tid]);
    }
}

__global__ __launch_bounds__(256) void finalize_kernel(const double* __restrict__ rowK,
                                                       const double* __restrict__ rowL,
                                                       const Scalars* __restrict__ sc,
                                                       float* __restrict__ out) {
    __shared__ double red[256];
    const int tid = threadIdx.x;
    double sk = 0.0, sl = 0.0, skl = 0.0, skk = 0.0, sll = 0.0;
    for (int i = tid; i < NPTS; i += 256) {
        double k = rowK[i], l = rowL[i];
        sk += k; sl += l; skl += k * l; skk += k * k; sll += l * l;
    }
    sk  = block_reduce256(sk, red);
    sl  = block_reduce256(sl, red);
    skl = block_reduce256(skl, red);
    skk = block_reduce256(skk, red);
    sll = block_reduce256(sll, red);
    if (tid == 0) {
        const double n = (double)NPTS;
        double hKL = sc->sumKL - (2.0 / n) * skl + (sk * sl) / (n * n);
        double hKK = sc->sumKK - (2.0 / n) * skk + (sk * sk) / (n * n);
        double hLL = sc->sumLL - (2.0 / n) * sll + (sl * sl) / (n * n);
        out[0] = (float)(hKL / sqrt(hKK * hLL));
    }
}

// ---------------- launcher ----------------

extern "C" void kernel_launch(void* const* d_in, const int* in_sizes, int n_in,
                              void* d_out, int out_size, void* d_ws, size_t ws_size,
                              hipStream_t stream) {
    const float* A = (const float*)d_in[0];
    const float* B = (const float*)d_in[1];
    float* out = (float*)d_out;

    char* ws = (char*)d_ws;
    const size_t BF_BYTES = (size_t)NPTS * DIM * sizeof(__bf16);   // 8 MB
    size_t off = 0;
    __bf16* Abf = (__bf16*)(ws + off); off += BF_BYTES;
    __bf16* Bbf = (__bf16*)(ws + off); off += BF_BYTES;
    float* sqA  = (float*)(ws + off);  off += NPTS * sizeof(float);
    float* sqB  = (float*)(ws + off);  off += NPTS * sizeof(float);
    unsigned* histA = (unsigned*)(ws + off); off += NB * sizeof(unsigned);
    unsigned* histB = (unsigned*)(ws + off); off += NB * sizeof(unsigned);
    double* rowK = (double*)(ws + off); off += NPTS * sizeof(double);
    double* rowL = (double*)(ws + off); off += NPTS * sizeof(double);
    Scalars* sc  = (Scalars*)(ws + off);

    init_ws_kernel<<<32, 256, 0, stream>>>(histA, histB, rowK, rowL, sc);
    convert_kernel<<<2048, 256, 0, stream>>>(A, B, Abf, Bbf);
    rownorm_kernel<<<dim3(NPTS, 2), 128, 0, stream>>>(A, B, sqA, sqB);

    dim3 grid(NPTS / TILE, NPTS / TILE);
    hist_kernel<<<grid, 256, 0, stream>>>(Abf, sqA, histA);
    hist_kernel<<<grid, 256, 0, stream>>>(Bbf, sqB, histB);
    median_kernel<<<1, 1, 0, stream>>>(histA, histB, sc);

    hsic_kernel<<<grid, 256, 0, stream>>>(Abf, Bbf, sqA, sqB, sc, rowK, rowL);
    finalize_kernel<<<1, 256, 0, stream>>>(rowK, rowL, sc, out);
}
